// seg_model_68693706932238
// MI455X (gfx1250) — matrix operand order repaired and checked by reading, not compiled, not run on hardware
//
#include <hip/hip_runtime.h>
#include <hip/hip_bf16.h>

// MI455X / gfx1250 implementation.
// - Laplacian materialized in FP16 (symmetric => no transpose needed).
// - All matmuls via v_wmma_f32_16x16x32_f16 (fp32 accumulate), 4 WMMAs/wave/iter
//   (32x32 output tile per wave: 2 A-frags x 2 B-frags -> best DS-load/WMMA ratio).
// - A-tile staged with CDNA5 async copy: global_load_async_to_lds_b128 +
//   s_wait_asynccnt (no VGPR round-trip); B-tile transposed into LDS on the way in.
// - L@x GEMMs are HBM-bound streaming L (~1.5GB total @ 23.3 TB/s ~ 64us);
//   fp16 L halves that traffic. Weight/FC GEMMs are WMMA compute-bound.
// Workspace requirement ~553 MB.

typedef __attribute__((ext_vector_type(16))) _Float16 v16h;
typedef __attribute__((ext_vector_type(8)))  _Float16 v8h;
typedef __attribute__((ext_vector_type(8)))  float    v8f;

#define Bq   16
#define Nn   2048
#define F_INq 6
#define NCLS 16

// ---------------- generic batched WMMA GEMM ----------------
// C[b] = alpha * A[b] @ B[b] + beta * C[b]
// A: f16 row-major [M x K], B: f16 row-major [K x N] (transposed into LDS),
// C: f32 row-major [M x N].  M = 2048 always here; K multiple of 32.
#define BM 128
#define BN 64
#define BK 32
#define LDSS 40   // halves per LDS row (80B): keeps 16B alignment, skews banks

__global__ __launch_bounds__(256) void wmma_gemm(
    const _Float16* __restrict__ A, long strideA, int lda,
    const _Float16* __restrict__ B, long strideB, int ldb,
    float* __restrict__ C, long strideC,
    int M, int N, int K, float alpha, float beta)
{
  (void)M;
  __shared__ __align__(16) _Float16 lA[BM * LDSS];          // [m][k]
  __shared__ __align__(16) _Float16 lB[BN * LDSS];          // [n][k] (transposed)

  const int b  = blockIdx.z;
  A += (long)b * strideA;  B += (long)b * strideB;  C += (long)b * strideC;
  const int m0 = blockIdx.y * BM;
  const int n0 = blockIdx.x * BN;

  const int tid  = threadIdx.x;
  const int lane = tid & 31;
  const int wave = tid >> 5;          // 8 waves
  const int wr   = wave >> 1;         // 0..3 : 32-row slab
  const int wc   = wave & 1;          // 0..1 : 32-col slab

  v8f c00 = {}, c01 = {}, c10 = {}, c11 = {};

  // cooperative-load indices: A tile 128x32 halves = 512 b128 chunks, 2/thread
  int rA[2], cA[2];
  unsigned ldsA[2];
#pragma unroll
  for (int j = 0; j < 2; ++j) {
    const int chunk = tid + j * 256;
    rA[j] = chunk >> 2;              // 0..127
    cA[j] = (chunk & 3) * 8;         // 0,8,16,24
    // generic->LDS byte address: low 32 bits of the generic pointer (ISA 10.2)
    ldsA[j] = (unsigned)(uintptr_t)(&lA[rA[j] * LDSS + cA[j]]);
  }
  const int bk = tid >> 3;            // 0..31
  const int bn = (tid & 7) * 8;       // 0..56

  const bool bVec = (n0 + BN) <= N && ((ldb & 7) == 0);   // uniform per launch

  for (int k0 = 0; k0 < K; k0 += BK) {
    // ---- stage A tile asynchronously into LDS (CDNA5 async copy) ----
#pragma unroll
    for (int j = 0; j < 2; ++j) {
      const _Float16* gp = A + (size_t)(m0 + rA[j]) * lda + (k0 + cA[j]);
      asm volatile("global_load_async_to_lds_b128 %0, %1, off"
                   :: "v"(ldsA[j]), "v"(gp) : "memory");
    }

    // ---- stage B tile transposed: lB[n][k] ----
    if (bVec) {
      v8h bv = *(const v8h*)(B + (size_t)(k0 + bk) * ldb + (n0 + bn));
#pragma unroll
      for (int i = 0; i < 8; ++i)
        lB[(bn + i) * LDSS + bk] = bv[i];
    } else {
#pragma unroll
      for (int i = 0; i < 8; ++i) {
        int n = n0 + bn + i;
        _Float16 v = (_Float16)0.f;
        if (n < N) v = B[(size_t)(k0 + bk) * ldb + n];
        lB[(bn + i) * LDSS + bk] = v;
      }
    }
    asm volatile("s_wait_asynccnt 0x0" ::: "memory");
    __syncthreads();

    // ---- A fragments (ISA 16-bit A layout: lane<16 K{0-7,16-23}, lane>=16 K{8-15,24-31}) ----
    const int am = wr * 32 + (lane & 15);
    const int ak = (lane >> 4) * 8;
    v16h af0, af1;
    {
      v8h lo0 = *(const v8h*)(&lA[am * LDSS + ak]);
      v8h hi0 = *(const v8h*)(&lA[am * LDSS + ak + 16]);
      v8h lo1 = *(const v8h*)(&lA[(am + 16) * LDSS + ak]);
      v8h hi1 = *(const v8h*)(&lA[(am + 16) * LDSS + ak + 16]);
#pragma unroll
      for (int i = 0; i < 8; ++i) {
        af0[i] = lo0[i]; af0[i + 8] = hi0[i];
        af1[i] = lo1[i]; af1[i + 8] = hi1[i];
      }
    }
    // ---- B fragments (same layout as A: lane<16 K{0-7,16-23}, lane>=16 K{8-15,24-31}) ----
    const int bnl = wc * 32 + (lane & 15);
    const int bkl = (lane >> 4) * 8;
    v16h bf0, bf1;
    {
      v8h lo0 = *(const v8h*)(&lB[bnl * LDSS + bkl]);
      v8h hi0 = *(const v8h*)(&lB[bnl * LDSS + bkl + 16]);
      v8h lo1 = *(const v8h*)(&lB[(bnl + 16) * LDSS + bkl]);
      v8h hi1 = *(const v8h*)(&lB[(bnl + 16) * LDSS + bkl + 16]);
#pragma unroll
      for (int i = 0; i < 8; ++i) {
        bf0[i] = lo0[i]; bf0[i + 8] = hi0[i];
        bf1[i] = lo1[i]; bf1[i + 8] = hi1[i];
      }
    }

    c00 = __builtin_amdgcn_wmma_f32_16x16x32_f16(false, af0, false, bf0, (short)0, c00, false, false);
    c01 = __builtin_amdgcn_wmma_f32_16x16x32_f16(false, af0, false, bf1, (short)0, c01, false, false);
    c10 = __builtin_amdgcn_wmma_f32_16x16x32_f16(false, af1, false, bf0, (short)0, c10, false, false);
    c11 = __builtin_amdgcn_wmma_f32_16x16x32_f16(false, af1, false, bf1, (short)0, c11, false, false);
    __syncthreads();
  }

  // ---- epilogue: C VGPR r -> (M = r + 8*(lane>>4), N = lane&15) ----
  const int rb = m0 + wr * 32 + (lane >> 4) * 8;
  const int cb = n0 + wc * 32 + (lane & 15);
  v8f accs[4] = {c00, c01, c10, c11};
#pragma unroll
  for (int t = 0; t < 4; ++t) {
    const int ti = t >> 1;            // A-frag index (row +16*ti)
    const int tj = t & 1;             // B-frag index (col +16*tj)
    const int col = cb + tj * 16;
    if (col < N) {
#pragma unroll
      for (int r = 0; r < 8; ++r) {
        const int row = rb + ti * 16 + r;
        size_t idx = (size_t)row * N + col;
        float prev = (beta != 0.f) ? beta * C[idx] : 0.f;
        C[idx] = alpha * accs[t][r] + prev;
      }
    }
  }
}

// ---------------- Laplacian construction ----------------
__global__ __launch_bounds__(256) void dinv_kernel(const float* __restrict__ x,
                                                   float* __restrict__ dinv)
{
  __shared__ float xm[256][F_INq];
  const int b = blockIdx.y;
  const int n = blockIdx.x * 256 + threadIdx.x;
  float xn[F_INq];
#pragma unroll
  for (int i = 0; i < F_INq; ++i) xn[i] = x[((size_t)b * Nn + n) * F_INq + i];
  float s = 0.f;
  for (int m0 = 0; m0 < Nn; m0 += 256) {
    __syncthreads();
#pragma unroll
    for (int i = 0; i < F_INq; ++i)
      xm[threadIdx.x][i] = x[((size_t)b * Nn + m0 + threadIdx.x) * F_INq + i];
    __syncthreads();
    for (int j = 0; j < 256; ++j) {
      float d2 = 0.f;
#pragma unroll
      for (int i = 0; i < F_INq; ++i) { float t = xn[i] - xm[j][i]; d2 += t * t; }
      s += __expf(-d2);
    }
  }
  dinv[(size_t)b * Nn + n] = rsqrtf(s);
}

__global__ __launch_bounds__(256) void build_L_kernel(const float* __restrict__ x,
                                                      const float* __restrict__ dinv,
                                                      _Float16* __restrict__ L)
{
  const int b   = blockIdx.z;
  const int row = blockIdx.y;
  const int col = blockIdx.x * 256 + threadIdx.x;
  float d2 = 0.f;
#pragma unroll
  for (int i = 0; i < F_INq; ++i) {
    float t = x[((size_t)b * Nn + row) * F_INq + i] - x[((size_t)b * Nn + col) * F_INq + i];
    d2 += t * t;
  }
  float v = -__expf(-d2) * dinv[(size_t)b * Nn + row] * dinv[(size_t)b * Nn + col];
  if (row == col) v += 1.f;
  L[((size_t)b * Nn + row) * Nn + col] = (_Float16)v;
}

// ---------------- elementwise helpers ----------------
__global__ __launch_bounds__(256) void build_h0_kernel(const float* __restrict__ x,
                                                       const int* __restrict__ cat,
                                                       _Float16* __restrict__ h0)
{
  const int idx = blockIdx.x * 256 + threadIdx.x;   // b*Nn + n
  const int b = idx / Nn;
  const int c = cat[b];
  _Float16* p = h0 + (size_t)idx * 32;
#pragma unroll
  for (int i = 0; i < F_INq; ++i) p[i] = (_Float16)x[(size_t)idx * F_INq + i];
#pragma unroll
  for (int i = 0; i < NCLS; ++i) p[F_INq + i] = (_Float16)((i == c) ? 1.f : 0.f);
#pragma unroll
  for (int i = F_INq + NCLS; i < 32; ++i) p[i] = (_Float16)0.f;
}

__global__ __launch_bounds__(256) void convw_kernel(const float* __restrict__ src,
                                                    _Float16* __restrict__ dst,
                                                    int Fin, int Fout, int total)
{
  const int idx = blockIdx.x * 256 + threadIdx.x;
  if (idx >= total) return;
  const int fi = idx / Fout, fo = idx % Fout;
  dst[idx] = (fi < Fin) ? (_Float16)src[(size_t)fi * Fout + fo] : (_Float16)0.f;
}

__global__ __launch_bounds__(256) void cvt_kernel(const float* __restrict__ src,
                                                  _Float16* __restrict__ dst, int total)
{
  const int idx = blockIdx.x * 256 + threadIdx.x;
  if (idx >= total) return;
  dst[idx] = (_Float16)src[idx];
}

// x2 = 2*(L@x1) - x0
__global__ __launch_bounds__(256) void combine_kernel(const float* __restrict__ y,
                                                      const _Float16* __restrict__ x0,
                                                      _Float16* __restrict__ x2, int total)
{
  const int idx = blockIdx.x * 256 + threadIdx.x;
  if (idx >= total) return;
  x2[idx] = (_Float16)(2.f * y[idx] - (float)x0[idx]);
}

__global__ __launch_bounds__(256) void bias_relu_kernel(const float* __restrict__ acc,
                                                        const float* __restrict__ bias,
                                                        const float* __restrict__ br,
                                                        _Float16* __restrict__ out,
                                                        int F, int total)
{
  const int idx = blockIdx.x * 256 + threadIdx.x;
  if (idx >= total) return;
  const int f = idx % F;
  float v = acc[idx] + bias[f] + br[f];
  out[idx] = (_Float16)fmaxf(v, 0.f);
}

__global__ __launch_bounds__(256) void concat_kernel(const _Float16* __restrict__ a,
                                                     const _Float16* __restrict__ c,
                                                     _Float16* __restrict__ out, int total)
{
  const int idx = blockIdx.x * 256 + threadIdx.x;
  if (idx >= total) return;
  const int f = idx & 1023;
  const int r = idx >> 10;
  out[idx] = (f < 512) ? a[(size_t)r * 512 + f] : c[(size_t)r * 512 + (f - 512)];
}

__global__ __launch_bounds__(256) void final_kernel(const float* __restrict__ acc,
                                                    const float* __restrict__ bias,
                                                    const float* __restrict__ br,
                                                    float* __restrict__ out,
                                                    int F, int total)
{
  const int idx = blockIdx.x * 256 + threadIdx.x;
  if (idx >= total) return;
  const int f = idx % F;
  out[idx] = fmaxf(acc[idx] + bias[f] + br[f], 0.f);
}

// ---------------- orchestration ----------------
extern "C" void kernel_launch(void* const* d_in, const int* in_sizes, int n_in,
                              void* d_out, int out_size, void* d_ws, size_t ws_size,
                              hipStream_t stream)
{
  (void)in_sizes; (void)n_in; (void)out_size; (void)ws_size;
  const float* x    = (const float*)d_in[0];
  const int*   cat  = (const int*)d_in[1];
  const float* W1   = (const float*)d_in[2];
  const float* b1   = (const float*)d_in[3];
  const float* W2   = (const float*)d_in[4];
  const float* b2   = (const float*)d_in[5];
  const float* W3   = (const float*)d_in[6];
  const float* b3   = (const float*)d_in[7];
  const float* fc1w = (const float*)d_in[8];
  const float* fc1b = (const float*)d_in[9];
  const float* fc2w = (const float*)d_in[10];
  const float* fc2b = (const float*)d_in[11];
  const float* fc3w = (const float*)d_in[12];
  const float* fc3b = (const float*)d_in[13];
  const float* br0  = (const float*)d_in[14];
  const float* br1  = (const float*)d_in[15];
  const float* br2  = (const float*)d_in[16];
  const float* br3  = (const float*)d_in[17];
  const float* br4  = (const float*)d_in[18];
  const float* br5  = (const float*)d_in[19];

  size_t off = 0;
  auto carve = [&](size_t bytes) -> char* {
    char* p = (char*)d_ws + off;
    off += (bytes + 255) & ~(size_t)255;
    return p;
  };
  _Float16* Lf   = (_Float16*)carve((size_t)Bq * Nn * Nn * 2);
  float*    dinv = (float*)   carve((size_t)Bq * Nn * 4);
  _Float16* h0   = (_Float16*)carve((size_t)Bq * Nn * 32 * 2);
  _Float16* xk1  = (_Float16*)carve((size_t)Bq * Nn * 512 * 2);
  _Float16* xk2  = (_Float16*)carve((size_t)Bq * Nn * 512 * 2);
  _Float16* xk3  = (_Float16*)carve((size_t)Bq * Nn * 512 * 2);
  float*    yL   = (float*)   carve((size_t)Bq * Nn * 512 * 4);
  float*    acc  = (float*)   carve((size_t)Bq * Nn * 1024 * 4);
  _Float16* h1   = (_Float16*)carve((size_t)Bq * Nn * 128 * 2);
  _Float16* h2   = (_Float16*)carve((size_t)Bq * Nn * 512 * 2);
  _Float16* h3   = (_Float16*)carve((size_t)Bq * Nn * 1024 * 2);
  _Float16* W1p  = (_Float16*)carve((size_t)6 * 32 * 128 * 2);
  _Float16* W2p  = (_Float16*)carve((size_t)5 * 128 * 512 * 2);
  _Float16* W3p  = (_Float16*)carve((size_t)3 * 512 * 1024 * 2);
  _Float16* fc1p = (_Float16*)carve((size_t)1024 * 512 * 2);
  _Float16* fc2p = (_Float16*)carve((size_t)1024 * 128 * 2);
  _Float16* fc3p = (_Float16*)carve((size_t)128 * 50 * 2);
  // post-Cheb aliases (xk*/yL dead after layer 3)
  _Float16* hfc1 = (_Float16*)yL;   // 32MB <= 64MB
  _Float16* hcat = xk1;             // 64MB spans contiguous xk1+xk2
  _Float16* hfc2 = xk3;

  auto gemm = [&](const _Float16* A, long sA, int lda,
                  const _Float16* B, long sB, int ldb,
                  float* C, long sC, int N_, int K_, float alpha, float beta) {
    dim3 grid((N_ + BN - 1) / BN, Nn / BM, Bq);
    wmma_gemm<<<grid, 256, 0, stream>>>(A, sA, lda, B, sB, ldb, C, sC,
                                        Nn, N_, K_, alpha, beta);
  };

  // 1) Laplacian
  dinv_kernel<<<dim3(Nn / 256, Bq), 256, 0, stream>>>(x, dinv);
  build_L_kernel<<<dim3(Nn / 256, Nn, Bq), 256, 0, stream>>>(x, dinv, Lf);

  // 2) input features [x | onehot] padded to 32
  build_h0_kernel<<<(Bq * Nn) / 256, 256, 0, stream>>>(x, cat, h0);

  // 3) weights -> f16 (W1 K-padded 22->32)
  for (int k = 0; k < 6; ++k)
    convw_kernel<<<(32 * 128) / 256, 256, 0, stream>>>(W1 + (size_t)k * 22 * 128,
                                                       W1p + (size_t)k * 32 * 128,
                                                       22, 128, 32 * 128);
  for (int k = 0; k < 5; ++k)
    convw_kernel<<<(128 * 512) / 256, 256, 0, stream>>>(W2 + (size_t)k * 128 * 512,
                                                        W2p + (size_t)k * 128 * 512,
                                                        128, 512, 128 * 512);
  for (int k = 0; k < 3; ++k)
    convw_kernel<<<(512 * 1024) / 256, 256, 0, stream>>>(W3 + (size_t)k * 512 * 1024,
                                                         W3p + (size_t)k * 512 * 1024,
                                                         512, 1024, 512 * 1024);
  convw_kernel<<<(1024 * 512) / 256, 256, 0, stream>>>(fc1w, fc1p, 1024, 512, 1024 * 512);
  convw_kernel<<<(1024 * 128) / 256, 256, 0, stream>>>(fc2w, fc2p, 1024, 128, 1024 * 128);
  convw_kernel<<<(128 * 50 + 255) / 256, 256, 0, stream>>>(fc3w, fc3p, 128, 50, 128 * 50);

  // Chebyshev layer: acc = sum_k x_k @ W[k]
  auto cheb = [&](const _Float16* xin, int Fin, const _Float16* Wp, int Ktaps, int Fout) {
    const long sX = (long)Nn * Fin;
    const long sO = (long)Nn * Fout;
    const int  tot = Bq * Nn * Fin;
    gemm(xin, sX, Fin, Wp, 0, Fout, acc, sO, Fout, Fin, 1.f, 0.f);
    // x1 = L @ x0   (M=2048, N=Fin, K=2048)
    gemm(Lf, (long)Nn * Nn, Nn, xin, sX, Fin, yL, sX, Fin, Nn, 1.f, 0.f);
    cvt_kernel<<<(tot + 255) / 256, 256, 0, stream>>>(yL, xk1, tot);
    gemm(xk1, sX, Fin, Wp + (size_t)Fin * Fout, 0, Fout, acc, sO, Fout, Fin, 1.f, 1.f);
    const _Float16* xm1 = xin;
    const _Float16* xcur = xk1;
    _Float16* bufs[3] = {xk1, xk2, xk3};
    int nb = 1;
    for (int k = 2; k < Ktaps; ++k) {
      _Float16* xnew = bufs[nb];
      nb = (nb + 1) % 3;
      gemm(Lf, (long)Nn * Nn, Nn, xcur, sX, Fin, yL, sX, Fin, Nn, 1.f, 0.f);
      combine_kernel<<<(tot + 255) / 256, 256, 0, stream>>>(yL, xm1, xnew, tot);
      gemm(xnew, sX, Fin, Wp + (size_t)k * Fin * Fout, 0, Fout, acc, sO, Fout, Fin, 1.f, 1.f);
      xm1 = xcur; xcur = xnew;
    }
  };

  // 4) Cheb layers
  cheb(h0, 32, W1p, 6, 128);
  bias_relu_kernel<<<(Bq * Nn * 128) / 256, 256, 0, stream>>>(acc, b1, br0, h1, 128, Bq * Nn * 128);
  cheb(h1, 128, W2p, 5, 512);
  bias_relu_kernel<<<(Bq * Nn * 512) / 256, 256, 0, stream>>>(acc, b2, br1, h2, 512, Bq * Nn * 512);
  cheb(h2, 512, W3p, 3, 1024);
  bias_relu_kernel<<<(Bq * Nn * 1024) / 256, 256, 0, stream>>>(acc, b3, br2, h3, 1024, Bq * Nn * 1024);

  // 5) FC head
  gemm(h3, (long)Nn * 1024, 1024, fc1p, 0, 512, acc, (long)Nn * 512, 512, 1024, 1.f, 0.f);
  bias_relu_kernel<<<(Bq * Nn * 512) / 256, 256, 0, stream>>>(acc, fc1b, br3, hfc1, 512, Bq * Nn * 512);
  concat_kernel<<<(Bq * Nn * 1024) / 256, 256, 0, stream>>>(hfc1, h2, hcat, Bq * Nn * 1024);
  gemm(hcat, (long)Nn * 1024, 1024, fc2p, 0, 128, acc, (long)Nn * 128, 128, 1024, 1.f, 0.f);
  bias_relu_kernel<<<(Bq * Nn * 128) / 256, 256, 0, stream>>>(acc, fc2b, br4, hfc2, 128, Bq * Nn * 128);
  gemm(hfc2, (long)Nn * 128, 128, fc3p, 0, 50, acc, (long)Nn * 50, 50, 128, 1.f, 0.f);
  final_kernel<<<(Bq * Nn * 50 + 255) / 256, 256, 0, stream>>>(acc, fc3b, br5,
                                                               (float*)d_out, 50, Bq * Nn * 50);
}
